// Attention_12369505813001
// MI455X (gfx1250) — matrix operand order repaired and checked by reading, not compiled, not run on hardware
//
#include <hip/hip_runtime.h>

// ---------------------------------------------------------------------------
// Attention (Luong-style) for MI455X / gfx1250, f16 WMMA pipeline.
// B=32, T=512, S=1024, H=1024
// Round 2: GEMM waves re-tiled to 32x64 (2x4 WMMA tiles) to raise operand
// reuse per WMMA (0.75 KB/WMMA vs 1.25 KB/WMMA) — the matrix pipe is
// vector-memory-return bound otherwise.
// ---------------------------------------------------------------------------

#define B_ 32
#define T_ 512
#define S_ 1024
#define H_ 1024

typedef __attribute__((ext_vector_type(16))) _Float16 v16h;
typedef __attribute__((ext_vector_type(8)))  _Float16 v8h;
typedef __attribute__((ext_vector_type(4)))  _Float16 v4h;
typedef __attribute__((ext_vector_type(8)))  float    v8f;

__device__ __forceinline__ v8f wmma_f16(v16h a, v16h b, v8f c) {
  // D = A(16x32 f16) * B(32x16 f16) + C(16x16 f32)
  return __builtin_amdgcn_wmma_f32_16x16x32_f16(
      /*neg_a=*/false, a, /*neg_b=*/false, b,
      /*c_mod=*/(short)0, c, /*reuse_a=*/false, /*reuse_b=*/false);
}

__device__ __forceinline__ v8f zero8() {
  v8f z;
#pragma unroll
  for (int i = 0; i < 8; ++i) z[i] = 0.0f;
  return z;
}

__device__ __forceinline__ v16h make_afrag(const _Float16* p) {
  // Per-lane A fragment: 16 contiguous f16 at p (contiguous K layout:
  // K=sel*16..+15, matching B fragments).
  v8h lo = *(const v8h*)p;
  v8h hi = *(const v8h*)(p + 8);
  v16h a;
#pragma unroll
  for (int i = 0; i < 8; ++i) { a[i] = lo[i]; a[8 + i] = hi[i]; }
  return a;
}

// ---------------------------------------------------------------------------
// 1) f32 -> f16 convert (vectorized x4)
// ---------------------------------------------------------------------------
__global__ __launch_bounds__(256) void cvt_f32_f16(
    const float* __restrict__ src, _Float16* __restrict__ dst, int n4) {
  int i = blockIdx.x * 256 + threadIdx.x;
  if (i < n4) {
    float4 v = ((const float4*)src)[i];
    v4h h;
    h[0] = (_Float16)v.x; h[1] = (_Float16)v.y;
    h[2] = (_Float16)v.z; h[3] = (_Float16)v.w;
    ((v4h*)dst)[i] = h;
  }
}

// ---------------------------------------------------------------------------
// 2) LDS-tiled transpose+convert: Enc f32 [B,S,H] -> EncT f16 [B,H,S]
// ---------------------------------------------------------------------------
__global__ __launch_bounds__(256) void transpose_cvt(
    const float* __restrict__ src, _Float16* __restrict__ dst) {
  __shared__ _Float16 tile[32][33];
  const int bz = blockIdx.z;
  const int s0 = blockIdx.x * 32, h0 = blockIdx.y * 32;
  const int tx = threadIdx.x, ty = threadIdx.y;  // 32 x 8
  const float* sp = src + (size_t)bz * S_ * H_;
#pragma unroll
  for (int j = 0; j < 4; ++j)
    tile[ty + 8 * j][tx] = (_Float16)sp[(size_t)(s0 + ty + 8 * j) * H_ + h0 + tx];
  __syncthreads();
  _Float16* dp = dst + (size_t)bz * H_ * S_;
#pragma unroll
  for (int j = 0; j < 4; ++j)
    dp[(size_t)(h0 + ty + 8 * j) * S_ + s0 + tx] = tile[tx][ty + 8 * j];
}

// ---------------------------------------------------------------------------
// 3) Batched NT WMMA GEMM: C[M,N] f16 = A[M,K] f16 * B[N,K]^T f16
//    Block tile 128(M) x 128(N), 8 waves (4 M-strips x 2 N-strips),
//    each wave 32x64 = 2x4 accum tiles, B fragment reused by 2 A fragments.
// ---------------------------------------------------------------------------
__global__ __launch_bounds__(256) void gemm_nt_f16(
    const _Float16* __restrict__ A, const _Float16* __restrict__ Bm,
    _Float16* __restrict__ C, int N, int K, int lda, int ldb,
    long long strideA, long long strideB, long long strideC) {
  const int lane = threadIdx.x & 31;
  const int wave = threadIdx.x >> 5;
  const int wm = wave & 3, wn = wave >> 2;
  const int lh = lane & 15, sel = lane >> 4;
  A  += (long long)blockIdx.z * strideA;
  Bm += (long long)blockIdx.z * strideB;
  C  += (long long)blockIdx.z * strideC;
  const int m0 = blockIdx.x * 128 + wm * 32;
  const int n0 = blockIdx.y * 128 + wn * 64;

  v8f acc[2][4];
#pragma unroll
  for (int mi = 0; mi < 2; ++mi)
#pragma unroll
    for (int t = 0; t < 4; ++t) acc[mi][t] = zero8();

  const _Float16* aptr0 = A + (size_t)(m0 + lh) * lda + sel * 16;
  const _Float16* aptr1 = aptr0 + (size_t)16 * lda;
  for (int k = 0; k < K; k += 32) {
    v16h a0 = make_afrag(aptr0 + k);
    v16h a1 = make_afrag(aptr1 + k);
#pragma unroll
    for (int t = 0; t < 4; ++t) {
      const _Float16* bp = Bm + (size_t)(n0 + t * 16 + lh) * ldb + k + sel * 16;
      v16h bf = *(const v16h*)bp;
      acc[0][t] = wmma_f16(a0, bf, acc[0][t]);
      acc[1][t] = wmma_f16(a1, bf, acc[1][t]);
    }
  }
#pragma unroll
  for (int mi = 0; mi < 2; ++mi) {
#pragma unroll
    for (int t = 0; t < 4; ++t) {
      const int col = n0 + t * 16 + lh;
#pragma unroll
      for (int r = 0; r < 8; ++r) {
        const int row = m0 + mi * 16 + r + sel * 8;
        C[(size_t)row * N + col] = (_Float16)acc[mi][t][r];
      }
    }
  }
}

// ---------------------------------------------------------------------------
// 4) Fused scores + mask + softmax.
//    One WG (4 waves) per (batch, 16-row T-tile). Wave w owns S columns
//    [w*256, w*256+256) as 16 register-resident 16x16 f32 accum tiles.
// ---------------------------------------------------------------------------
__global__ __launch_bounds__(128) void attn_scores_softmax(
    const _Float16* __restrict__ QW, const _Float16* __restrict__ Enc,
    const long long* __restrict__ Len, _Float16* __restrict__ P) {
  const int b = blockIdx.y;
  const int t0 = blockIdx.x * 16;
  const int lane = threadIdx.x & 31;
  const int wave = threadIdx.x >> 5;  // 0..3
  const int lh = lane & 15, sel = lane >> 4;
  const int sbase = wave * 256;

  __shared__ float redMax[4][16];
  __shared__ float redSum[4][16];

  v8f sc[16];
#pragma unroll
  for (int t = 0; t < 16; ++t) sc[t] = zero8();

  const _Float16* aptr = QW + ((size_t)b * T_ + t0 + lh) * H_ + sel * 16;
  const _Float16* eb = Enc + (size_t)b * S_ * H_;
  for (int k = 0; k < H_; k += 32) {
    v16h a = make_afrag(aptr + k);
#pragma unroll
    for (int t = 0; t < 16; ++t) {
      const _Float16* bp = eb + (size_t)(sbase + t * 16 + lh) * H_ + k + sel * 16;
      v16h bf = *(const v16h*)bp;
      sc[t] = wmma_f16(a, bf, sc[t]);
    }
  }

  // mask: column s >= length -> -inf (column index is per-lane uniform)
  const int L = (int)Len[b];
  const float NEG_INF = -__builtin_inff();
#pragma unroll
  for (int t = 0; t < 16; ++t) {
    const int s = sbase + t * 16 + lh;
    if (s >= L) {
#pragma unroll
      for (int r = 0; r < 8; ++r) sc[t][r] = NEG_INF;
    }
  }

  // row max: in-lane over 16 tiles, xor-shuffle over the 16-lane half,
  // then cross-wave via LDS.
  float rmax[8];
#pragma unroll
  for (int r = 0; r < 8; ++r) {
    float m = sc[0][r];
#pragma unroll
    for (int t = 1; t < 16; ++t) m = fmaxf(m, sc[t][r]);
#pragma unroll
    for (int off = 1; off < 16; off <<= 1) m = fmaxf(m, __shfl_xor(m, off, 32));
    rmax[r] = m;
  }
  if (lh == 0) {
#pragma unroll
    for (int r = 0; r < 8; ++r) redMax[wave][r + sel * 8] = rmax[r];
  }
  __syncthreads();
#pragma unroll
  for (int r = 0; r < 8; ++r) {
    float m = redMax[0][r + sel * 8];
#pragma unroll
    for (int w = 1; w < 4; ++w) m = fmaxf(m, redMax[w][r + sel * 8]);
    rmax[r] = m;
  }

  // exp + row sum
  float rsum[8];
#pragma unroll
  for (int r = 0; r < 8; ++r) rsum[r] = 0.0f;
#pragma unroll
  for (int t = 0; t < 16; ++t) {
#pragma unroll
    for (int r = 0; r < 8; ++r) {
      float e = __expf(sc[t][r] - rmax[r]);
      sc[t][r] = e;
      rsum[r] += e;
    }
  }
#pragma unroll
  for (int r = 0; r < 8; ++r) {
#pragma unroll
    for (int off = 1; off < 16; off <<= 1) rsum[r] += __shfl_xor(rsum[r], off, 32);
  }
  if (lh == 0) {
#pragma unroll
    for (int r = 0; r < 8; ++r) redSum[wave][r + sel * 8] = rsum[r];
  }
  __syncthreads();
#pragma unroll
  for (int r = 0; r < 8; ++r) {
    float s = redSum[0][r + sel * 8];
#pragma unroll
    for (int w = 1; w < 4; ++w) s += redSum[w][r + sel * 8];
    rsum[r] = 1.0f / s;
  }

  // write probabilities (f16)
#pragma unroll
  for (int t = 0; t < 16; ++t) {
    const int s = sbase + t * 16 + lh;
#pragma unroll
    for (int r = 0; r < 8; ++r) {
      const int row = t0 + r + sel * 8;
      P[((size_t)b * T_ + row) * S_ + s] = (_Float16)(sc[t][r] * rsum[r]);
    }
  }
}

// ---------------------------------------------------------------------------
// 5) Output GEMM: out = tanh([Q | C] * W_out^T + b_out), f32 result.
//    Two K-phases over the concat dimension; 32x64 wave tiles.
// ---------------------------------------------------------------------------
__global__ __launch_bounds__(256) void out_gemm_tanh(
    const _Float16* __restrict__ Aq, const _Float16* __restrict__ Ac,
    const _Float16* __restrict__ Wo, const float* __restrict__ bias,
    float* __restrict__ Out) {
  const int lane = threadIdx.x & 31;
  const int wave = threadIdx.x >> 5;
  const int wm = wave & 3, wn = wave >> 2;
  const int lh = lane & 15, sel = lane >> 4;
  const int m0 = blockIdx.x * 128 + wm * 32;
  const int n0 = blockIdx.y * 128 + wn * 64;
  const int ldb = 2 * H_;

  v8f acc[2][4];
#pragma unroll
  for (int mi = 0; mi < 2; ++mi)
#pragma unroll
    for (int t = 0; t < 4; ++t) acc[mi][t] = zero8();

  const _Float16* aq0 = Aq + (size_t)(m0 + lh) * H_ + sel * 16;
  const _Float16* aq1 = aq0 + (size_t)16 * H_;
  const _Float16* ac0 = Ac + (size_t)(m0 + lh) * H_ + sel * 16;
  const _Float16* ac1 = ac0 + (size_t)16 * H_;

  for (int k = 0; k < H_; k += 32) {           // phase 1: query x W_out[:, :H]
    v16h a0 = make_afrag(aq0 + k);
    v16h a1 = make_afrag(aq1 + k);
#pragma unroll
    for (int t = 0; t < 4; ++t) {
      const _Float16* bp = Wo + (size_t)(n0 + t * 16 + lh) * ldb + k + sel * 16;
      v16h bf = *(const v16h*)bp;
      acc[0][t] = wmma_f16(a0, bf, acc[0][t]);
      acc[1][t] = wmma_f16(a1, bf, acc[1][t]);
    }
  }
  for (int k = 0; k < H_; k += 32) {           // phase 2: context x W_out[:, H:]
    v16h a0 = make_afrag(ac0 + k);
    v16h a1 = make_afrag(ac1 + k);
#pragma unroll
    for (int t = 0; t < 4; ++t) {
      const _Float16* bp = Wo + (size_t)(n0 + t * 16 + lh) * ldb + H_ + k + sel * 16;
      v16h bf = *(const v16h*)bp;
      acc[0][t] = wmma_f16(a0, bf, acc[0][t]);
      acc[1][t] = wmma_f16(a1, bf, acc[1][t]);
    }
  }

#pragma unroll
  for (int mi = 0; mi < 2; ++mi) {
#pragma unroll
    for (int t = 0; t < 4; ++t) {
      const int col = n0 + t * 16 + lh;
      const float bv = bias[col];
#pragma unroll
      for (int r = 0; r < 8; ++r) {
        const int row = m0 + mi * 16 + r + sel * 8;
        Out[(size_t)row * H_ + col] = tanhf(acc[mi][t][r] + bv);
      }
    }
  }
}

// ---------------------------------------------------------------------------
// Host-side launcher
// ---------------------------------------------------------------------------
extern "C" void kernel_launch(void* const* d_in, const int* in_sizes, int n_in,
                              void* d_out, int out_size, void* d_ws, size_t ws_size,
                              hipStream_t stream) {
  (void)in_sizes; (void)n_in; (void)out_size; (void)ws_size;

  const float*     query = (const float*)d_in[0];      // [B,T,H]
  const float*     enc   = (const float*)d_in[1];      // [B,S,H]
  const long long* lens  = (const long long*)d_in[2];  // [B] int64
  const float*     W_in  = (const float*)d_in[3];      // [H,H]
  const float*     W_out = (const float*)d_in[4];      // [H,2H]
  const float*     b_out = (const float*)d_in[5];      // [H]
  float*           out   = (float*)d_out;              // [B,T,H]

  const size_t BTH = (size_t)B_ * T_ * H_;     // 16,777,216
  const size_t BSH = (size_t)B_ * S_ * H_;     // 33,554,432
  const size_t HH  = (size_t)H_ * H_;          // 1,048,576
  const size_t H2H = (size_t)H_ * 2 * H_;      // 2,097,152
  const size_t BTS = (size_t)B_ * T_ * S_;     // 16,777,216

  _Float16* qB    = (_Float16*)d_ws;           // f16 query
  _Float16* eB    = qB + BTH;                  // f16 enc   [B,S,H]
  _Float16* eT    = eB + BSH;                  // f16 enc^T [B,H,S]
  _Float16* wInB  = eT + BSH;                  // f16 W_in
  _Float16* wOutB = wInB + HH;                 // f16 W_out
  _Float16* qwB   = wOutB + H2H;               // f16 qw    [B,T,H]
  _Float16* pB    = qwB + BTH;                 // f16 probs [B,T,S]
  _Float16* cB    = pB + BTS;                  // f16 ctx   [B,T,H]

  // 1) converts
  cvt_f32_f16<<<(int)(BTH / 4 + 255) / 256, 256, 0, stream>>>(query, qB, (int)(BTH / 4));
  cvt_f32_f16<<<(int)(BSH / 4 + 255) / 256, 256, 0, stream>>>(enc,   eB, (int)(BSH / 4));
  cvt_f32_f16<<<(int)(HH  / 4 + 255) / 256, 256, 0, stream>>>(W_in,  wInB, (int)(HH / 4));
  cvt_f32_f16<<<(int)(H2H / 4 + 255) / 256, 256, 0, stream>>>(W_out, wOutB, (int)(H2H / 4));

  // 2) enc transpose (for context GEMM in NT form)
  transpose_cvt<<<dim3(S_ / 32, H_ / 32, B_), dim3(32, 8), 0, stream>>>(enc, eT);

  // 3) qw = query * W_in^T   (M = B*T, N = H, K = H), block tile 128x128
  gemm_nt_f16<<<dim3((B_ * T_) / 128, H_ / 128, 1), 256, 0, stream>>>(
      qB, wInB, qwB, H_, H_, H_, H_, 0LL, 0LL, 0LL);

  // 4) scores + mask + softmax -> P
  attn_scores_softmax<<<dim3(T_ / 16, B_), 128, 0, stream>>>(qwB, eB, lens, pB);

  // 5) context = P * Enc, batched NT via Enc^T (M = T, N = H, K = S)
  gemm_nt_f16<<<dim3(T_ / 128, H_ / 128, B_), 256, 0, stream>>>(
      pB, eT, cB, H_, S_, S_, S_,
      (long long)T_ * S_, (long long)H_ * S_, (long long)T_ * H_);

  // 6) out = tanh([Q|C] * W_out^T + b)
  out_gemm_tanh<<<dim3((B_ * T_) / 128, H_ / 128), 256, 0, stream>>>(
      qB, cB, wOutB, b_out, out);
}
